// VisionMambaBlock_70179765616887
// MI455X (gfx1250) — matrix operand order repaired and checked by reading, not compiled, not run on hardware
//
#include <hip/hip_runtime.h>
#include <math.h>

// ---------------------------------------------------------------------------
// VisionMambaBlock for MI455X (gfx1250, wave32, WMMA f32_16x16x32_f16)
// B=2, H=W=64 -> TOK=8192 tokens, C=96, D_INNER=192, D_STATE=16, D_CONV=4
// ---------------------------------------------------------------------------

typedef __attribute__((ext_vector_type(16))) _Float16 v16h;
typedef __attribute__((ext_vector_type(8)))  float    v8f;

#define TOK   8192
#define SEQL  4096
#define NB    2
#define CDIM  96
#define DINN  192
#define NST   16
#define XPW   448   // xp panel width (448 = 14*32); live cols: 0..415

// ------------------------------- helpers -----------------------------------

__device__ __forceinline__ float wave_sum(float v) {
#pragma unroll
  for (int off = 16; off > 0; off >>= 1) v += __shfl_xor(v, off, 32);
  return v;
}

__device__ __forceinline__ float gelu_exact(float v) {
  return 0.5f * v * (1.0f + erff(v * 0.70710678118654752f));
}

// ------------------------------ f32 -> f16 ---------------------------------

__global__ void cvt_f16_kernel(const float* __restrict__ in,
                               _Float16* __restrict__ out, int n) {
  int i = blockIdx.x * blockDim.x + threadIdx.x;
  if (i < n) out[i] = (_Float16)in[i];
}

// --------------------------- LayerNorm over C=96 ---------------------------

__global__ void ln96_kernel(const float* __restrict__ in,
                            const float* __restrict__ g,
                            const float* __restrict__ b,
                            float* __restrict__ outF32,
                            _Float16* __restrict__ outF16, int ntok) {
  int gid  = blockIdx.x * blockDim.x + threadIdx.x;
  int tok  = gid >> 5;
  int lane = gid & 31;
  if (tok >= ntok) return;
  const float* row = in + (size_t)tok * CDIM;
  float x0 = row[lane], x1 = row[lane + 32], x2 = row[lane + 64];
  float mu = wave_sum(x0 + x1 + x2) * (1.0f / 96.0f);
  float d0 = x0 - mu, d1 = x1 - mu, d2 = x2 - mu;
  float var  = wave_sum(d0 * d0 + d1 * d1 + d2 * d2) * (1.0f / 96.0f);
  float rstd = rsqrtf(var + 1e-5f);
  float y0 = d0 * rstd * g[lane]      + b[lane];
  float y1 = d1 * rstd * g[lane + 32] + b[lane + 32];
  float y2 = d2 * rstd * g[lane + 64] + b[lane + 64];
  size_t o = (size_t)tok * CDIM;
  if (outF32) { outF32[o + lane] = y0; outF32[o + lane + 32] = y1; outF32[o + lane + 64] = y2; }
  if (outF16) { outF16[o + lane] = (_Float16)y0; outF16[o + lane + 32] = (_Float16)y1; outF16[o + lane + 64] = (_Float16)y2; }
}

// ------------------------- generic WMMA f16 GEMM ---------------------------
// C[M,N] = A[M,K](f16) * B[K,N](f16)  (+bias, +GELU, +residual), fp32 accum.
// block = 256 threads = 8 waves; block tile 256(M) x 32(N); K step 32.
// Wave w owns a 32x32 patch: 2x2 subtiles -> 4 v_wmma per K-step with
// fragment reuse.  A tile staged via global_load_async_to_lds_b128 (ASYNCcnt);
// the LDS destination is derived from &As[][] (ptrtoint) so the compiler
// sees As as escaped/written and keeps the fragment ds_loads live.
// Requires: M%256==0, N%32==0, K%32==0, lda/ldb multiples of 8.

struct H16bits { uint4 lo, hi; };

__global__ __launch_bounds__(256) void gemm16_kernel(
    const _Float16* __restrict__ A, int lda,
    const _Float16* __restrict__ B, int ldb,
    float* __restrict__ Cf32, _Float16* __restrict__ Cf16, int ldc,
    const float* __restrict__ bias, const float* __restrict__ resid,
    int M, int N, int K, int act) {
  __shared__ _Float16 As[256][32];   // A tile, row-major         (16 KB)
  __shared__ _Float16 BsT[32][32];   // B tile transposed [n][k]  ( 2 KB)

  const int mBase = blockIdx.y * 256;
  const int nBase = blockIdx.x * 32;
  const int tid   = threadIdx.x;
  const int wave  = tid >> 5;
  const int lane  = tid & 31;

  v8f acc[2][2] = {};

  for (int k0 = 0; k0 < K; k0 += 32) {
    // ---- stage A tile: 256x32 halves = 1024 x 16B via async DMA to LDS ----
#pragma unroll
    for (int i = 0; i < 4; ++i) {
      int li = tid + i * 256;
      int r  = li >> 2;
      int kk = (li & 3) * 8;
      // low 32 bits of the generic address == wave-relative LDS byte offset;
      // the ptrtoint also escapes As so the async write is not optimized away
      unsigned lds_off = (unsigned)(size_t)&As[r][kk];
      unsigned long long ga =
          (unsigned long long)&A[(size_t)(mBase + r) * lda + k0 + kk];
      asm volatile("global_load_async_to_lds_b128 %0, %1, off"
                   :: "v"(lds_off), "v"(ga) : "memory");
    }
    // ---- stage B tile transposed (coalesced along n) ----
#pragma unroll
    for (int i = 0; i < 4; ++i) {
      int li = tid + i * 256;
      int kr = li >> 5;
      int nc = li & 31;
      BsT[nc][kr] = B[(size_t)(k0 + kr) * ldb + nBase + nc];
    }
    if (k0 + 32 < K)  // prefetch next A tile (global_prefetch_b8)
      __builtin_prefetch(&A[(size_t)(mBase + tid) * lda + k0 + 32], 0, 1);
    asm volatile("s_wait_asynccnt 0x0" ::: "memory");
    __syncthreads();

    // ---- gather fragments per CDNA5 16-bit WMMA VGPR layouts ----
    const int kh8  = (lane >> 4) * 8;    // A: K chunks of 8 per lane-half
    const int kh16 = (lane >> 4) * 16;   // unused; B also gathers with kh8
    v16h afrag[2], bfrag[2];
#pragma unroll
    for (int mi = 0; mi < 2; ++mi) {
      int mrow = wave * 32 + mi * 16 + (lane & 15);
      H16bits t;
      t.lo = *(const uint4*)&As[mrow][kh8];
      t.hi = *(const uint4*)&As[mrow][16 + kh8];
      afrag[mi] = __builtin_bit_cast(v16h, t);
    }
#pragma unroll
    for (int nj = 0; nj < 2; ++nj) {
      int ncol = nj * 16 + (lane & 15);
      H16bits t;
      t.lo = *(const uint4*)&BsT[ncol][kh8];
      t.hi = *(const uint4*)&BsT[ncol][16 + kh8];
      bfrag[nj] = __builtin_bit_cast(v16h, t);
    }
#pragma unroll
    for (int mi = 0; mi < 2; ++mi)
#pragma unroll
      for (int nj = 0; nj < 2; ++nj)
        acc[mi][nj] = __builtin_amdgcn_wmma_f32_16x16x32_f16(
            false, afrag[mi], false, bfrag[nj], (short)0, acc[mi][nj],
            false, false);
    __syncthreads();
  }

  // ---- epilogue: D layout -> row = +(lane>>4)*8 + r, col = lane&15 ----
#pragma unroll
  for (int mi = 0; mi < 2; ++mi) {
#pragma unroll
    for (int nj = 0; nj < 2; ++nj) {
      const int row0 = mBase + wave * 32 + mi * 16 + (lane >> 4) * 8;
      const int col  = nBase + nj * 16 + (lane & 15);
      const float bval = bias ? bias[col] : 0.0f;
#pragma unroll
      for (int r = 0; r < 8; ++r) {
        int row = row0 + r;
        float v = acc[mi][nj][r] + bval;
        if (act == 1) v = gelu_exact(v);
        if (resid) v += resid[(size_t)row * ldc + col];
        if (Cf32) Cf32[(size_t)row * ldc + col] = v;
        if (Cf16) Cf16[(size_t)row * ldc + col] = (_Float16)v;
      }
    }
  }
}

// --------------- depthwise causal conv (taps=4) + sigmoid(delta) -----------
// xp layout: [b, l, 448]: cols 0..191 x_conv, 192..383 delta, 384..399 B,
// 400..415 C, 416..447 padding (computed, unused)

__global__ void conv1d_sig_kernel(const float* __restrict__ xp,
                                  const float* __restrict__ cw,
                                  const float* __restrict__ cb,
                                  float* __restrict__ xc,
                                  float* __restrict__ dsig, int total) {
  int idx = blockIdx.x * blockDim.x + threadIdx.x;
  if (idx >= total) return;
  int d = idx % DINN;
  int l = (idx / DINN) % SEQL;
  int b = idx / (DINN * SEQL);
  const float* base = xp + (size_t)b * SEQL * XPW;
  float acc = cb[d];
#pragma unroll
  for (int k = 0; k < 4; ++k) {
    int ls  = l - 3 + k;
    float v = (ls >= 0) ? base[(size_t)ls * XPW + d] : 0.0f;
    acc += v * cw[d * 4 + k];
  }
  xc[idx] = acc;
  float dv  = base[(size_t)l * XPW + DINN + d];
  dsig[idx] = 1.0f / (1.0f + __expf(-dv));
}

// ------------------- bidirectional selective scan ---------------------------
// grid = B * D_INNER blocks, 512 threads = 16 waves (wave == state index n).
// Lane owns 128 contiguous steps; affine (A,B) chunk composition + shfl_up
// prefix scan (serial depth 128 + log2(32)); h*C reduced over 16 states into
// an LDS accumulator; y = acc + Dp*xc.

__global__ __launch_bounds__(512) void scan_kernel(
    const float* __restrict__ xp,     // Bm / Cm columns
    const float* __restrict__ dsig,   // sigmoid(delta)  [b,l,192]
    const float* __restrict__ xc,     // conv output     [b,l,192]
    const float* __restrict__ Amat,   // [16,16]
    const float* __restrict__ Dp,     // [192]
    _Float16* __restrict__ yF16) {    // [b,l,192] f16
  __shared__ float sDelta[SEQL];
  __shared__ float sXc[SEQL];
  __shared__ float sY[SEQL];

  const int b = blockIdx.x / DINN;
  const int d = blockIdx.x % DINN;
  const int tid  = threadIdx.x;
  const int n    = tid >> 5;
  const int lane = tid & 31;

  for (int i = tid; i < SEQL; i += 512) {
    size_t gi = ((size_t)b * SEQL + i) * DINN + d;
    sDelta[i] = dsig[gi];
    sXc[i]    = xc[gi];
    sY[i]     = 0.0f;
  }
  __syncthreads();

  float asum = 0.0f;
#pragma unroll
  for (int j = 0; j < NST; ++j) asum += Amat[n * NST + j];

  const float* xpB = xp + (size_t)b * SEQL * XPW + 2 * DINN + n;        // Bm
  const float* xpC = xp + (size_t)b * SEQL * XPW + 2 * DINN + NST + n;  // Cm

  for (int dir = 0; dir < 2; ++dir) {
    const int base = lane * 128;
    float Aseg = 1.0f, Bseg = 0.0f;
    for (int j = 0; j < 128; ++j) {
      int t = base + j;
      int l = dir ? (SEQL - 1 - t) : t;
      float dl = sDelta[l];
      float a  = __expf(dl * asum);
      float bu = dl * sXc[l] * xpB[(size_t)l * XPW];
      Bseg = a * Bseg + bu;
      Aseg *= a;
    }
    float Ai = Aseg, Bi = Bseg;
#pragma unroll
    for (int off = 1; off < 32; off <<= 1) {
      float pA = __shfl_up(Ai, (unsigned)off, 32);
      float pB = __shfl_up(Bi, (unsigned)off, 32);
      if (lane >= off) { Bi = Ai * pB + Bi; Ai = Ai * pA; }
    }
    float hin = __shfl_up(Bi, 1u, 32);
    if (lane == 0) hin = 0.0f;
    float h = hin;
    for (int j = 0; j < 128; ++j) {
      int t = base + j;
      int l = dir ? (SEQL - 1 - t) : t;
      float dl = sDelta[l];
      float a  = __expf(dl * asum);
      float bu = dl * sXc[l] * xpB[(size_t)l * XPW];
      h = a * h + bu;
      atomicAdd(&sY[l], h * xpC[(size_t)l * XPW]);
    }
  }
  __syncthreads();

  const float dpv = Dp[d];
  for (int i = tid; i < SEQL; i += 512) {
    float yv = sY[i] + dpv * sXc[i];
    yF16[((size_t)b * SEQL + i) * DINN + d] = (_Float16)yv;
  }
}

// ------------- cosine-similarity fusion -> cat = [f1 | fused] (f16) --------

__global__ void fuse_kernel(const float* __restrict__ f1,
                            const float* __restrict__ f2,
                            _Float16* __restrict__ cat, int ntok) {
  int gid  = blockIdx.x * blockDim.x + threadIdx.x;
  int tok  = gid >> 5;
  int lane = gid & 31;
  if (tok >= ntok) return;
  const float* r1 = f1 + (size_t)tok * CDIM;
  const float* r2 = f2 + (size_t)tok * CDIM;
  float a0 = r1[lane], a1 = r1[lane + 32], a2 = r1[lane + 64];
  float c0 = r2[lane], c1 = r2[lane + 32], c2 = r2[lane + 64];
  float s11 = wave_sum(a0 * a0 + a1 * a1 + a2 * a2);
  float s22 = wave_sum(c0 * c0 + c1 * c1 + c2 * c2);
  float s12 = wave_sum(a0 * c0 + a1 * c1 + a2 * c2);
  float n1 = fmaxf(sqrtf(s11), 1e-12f);
  float n2 = fmaxf(sqrtf(s22), 1e-12f);
  float sim = s12 / (n1 * n2);
  size_t o = (size_t)tok * (2 * CDIM);
  cat[o + lane]      = (_Float16)a0;
  cat[o + lane + 32] = (_Float16)a1;
  cat[o + lane + 64] = (_Float16)a2;
  cat[o + CDIM + lane]      = (_Float16)(sim * a0 + (1.0f - sim) * c0);
  cat[o + CDIM + lane + 32] = (_Float16)(sim * a1 + (1.0f - sim) * c1);
  cat[o + CDIM + lane + 64] = (_Float16)(sim * a2 + (1.0f - sim) * c2);
}

// ------------------- im2col for the 3x3 SAME conv (f16) --------------------

__global__ void im2col_kernel(const _Float16* __restrict__ cat,
                              _Float16* __restrict__ col, int total) {
  int idx = blockIdx.x * blockDim.x + threadIdx.x;
  if (idx >= total) return;
  int k = idx % 1728;
  int m = idx / 1728;
  int c = k % (2 * CDIM);
  int tap = k / (2 * CDIM);
  int ky = tap / 3 - 1, kx = tap % 3 - 1;
  int b = m >> 12;
  int hw = m & 4095;
  int h = hw >> 6, w = hw & 63;
  int hh = h + ky, ww = w + kx;
  _Float16 v = (_Float16)0.0f;
  if (hh >= 0 && hh < 64 && ww >= 0 && ww < 64)
    v = cat[(((size_t)b << 12) + hh * 64 + ww) * (2 * CDIM) + c];
  col[idx] = v;
}

// ------------------------------- host glue ---------------------------------

extern "C" void kernel_launch(void* const* d_in, const int* in_sizes, int n_in,
                              void* d_out, int out_size, void* d_ws, size_t ws_size,
                              hipStream_t stream) {
  const float* x      = (const float*)d_in[0];
  const float* norm_g = (const float*)d_in[1];
  const float* norm_b = (const float*)d_in[2];
  const float* Wp[2]  = {(const float*)d_in[3],  (const float*)d_in[11]};
  const float* cw[2]  = {(const float*)d_in[4],  (const float*)d_in[12]};
  const float* cb[2]  = {(const float*)d_in[5],  (const float*)d_in[13]};
  const float* Am[2]  = {(const float*)d_in[6],  (const float*)d_in[14]};
  const float* Dpp[2] = {(const float*)d_in[7],  (const float*)d_in[15]};
  const float* Wo[2]  = {(const float*)d_in[8],  (const float*)d_in[16]};
  const float* lng[2] = {(const float*)d_in[9],  (const float*)d_in[17]};
  const float* lnb[2] = {(const float*)d_in[10], (const float*)d_in[18]};
  const float* Wf = (const float*)d_in[19];
  const float* bf = (const float*)d_in[20];
  const float* W1 = (const float*)d_in[21];
  const float* b1 = (const float*)d_in[22];
  const float* W2 = (const float*)d_in[23];
  const float* b2 = (const float*)d_in[24];

  char* wp = (char*)d_ws;
  auto alloc = [&](size_t bytes) -> char* {
    char* r = wp;
    wp += (bytes + 255) & ~(size_t)255;
    return r;
  };

  _Float16* Wph[2]; _Float16* Woh[2];
  Wph[0] = (_Float16*)alloc((size_t)CDIM * 576 * 2);
  Wph[1] = (_Float16*)alloc((size_t)CDIM * 576 * 2);
  Woh[0] = (_Float16*)alloc((size_t)DINN * CDIM * 2);
  Woh[1] = (_Float16*)alloc((size_t)DINN * CDIM * 2);
  _Float16* Wfh  = (_Float16*)alloc((size_t)1728 * CDIM * 2);
  _Float16* W1h  = (_Float16*)alloc((size_t)CDIM * 384 * 2);
  _Float16* W2h  = (_Float16*)alloc((size_t)384 * CDIM * 2);
  _Float16* xnh  = (_Float16*)alloc((size_t)TOK * CDIM * 2);
  float*    xpB  = (float*)   alloc((size_t)TOK * XPW * 4);
  float*    xcB  = (float*)   alloc((size_t)TOK * DINN * 4);
  float*    dsB  = (float*)   alloc((size_t)TOK * DINN * 4);
  _Float16* yh   = (_Float16*)alloc((size_t)TOK * DINN * 2);
  float*    t96  = (float*)   alloc((size_t)TOK * CDIM * 4);
  float*    f1   = (float*)   alloc((size_t)TOK * CDIM * 4);
  float*    f2   = (float*)   alloc((size_t)TOK * CDIM * 4);
  _Float16* cath = (_Float16*)alloc((size_t)TOK * 2 * CDIM * 2);
  _Float16* colh = (_Float16*)alloc((size_t)TOK * 1728 * 2);
  _Float16* fch  = (_Float16*)alloc((size_t)TOK * CDIM * 2);
  _Float16* hidh = (_Float16*)alloc((size_t)TOK * 384 * 2);

  auto cvt = [&](const float* src, _Float16* dst, int n) {
    cvt_f16_kernel<<<(n + 255) / 256, 256, 0, stream>>>(src, dst, n);
  };
  cvt(Wp[0], Wph[0], CDIM * 576);
  cvt(Wp[1], Wph[1], CDIM * 576);
  cvt(Wo[0], Woh[0], DINN * CDIM);
  cvt(Wo[1], Woh[1], DINN * CDIM);
  cvt(Wf, Wfh, 1728 * CDIM);
  cvt(W1, W1h, CDIM * 384);
  cvt(W2, W2h, 384 * CDIM);

  // input layernorm -> f16
  ln96_kernel<<<(TOK * 32) / 256, 256, 0, stream>>>(
      x, norm_g, norm_b, nullptr, xnh, TOK);

  float* fout[2] = {f1, f2};
  for (int br = 0; br < 2; ++br) {
    // xp = xn @ Wp  (448-column live panel of the 576)
    gemm16_kernel<<<dim3(XPW / 32, TOK / 256), 256, 0, stream>>>(
        xnh, CDIM, Wph[br], 576, xpB, nullptr, XPW,
        nullptr, nullptr, TOK, XPW, CDIM, 0);
    // depthwise causal conv + sigmoid(delta)
    conv1d_sig_kernel<<<(TOK * DINN) / 256, 256, 0, stream>>>(
        xpB, cw[br], cb[br], xcB, dsB, TOK * DINN);
    // bidirectional selective scan -> y (f16)
    scan_kernel<<<NB * DINN, 512, 0, stream>>>(
        xpB, dsB, xcB, Am[br], Dpp[br], yh);
    // y @ Wo
    gemm16_kernel<<<dim3(CDIM / 32, TOK / 256), 256, 0, stream>>>(
        yh, DINN, Woh[br], CDIM, t96, nullptr, CDIM,
        nullptr, nullptr, TOK, CDIM, DINN, 0);
    // branch layernorm
    ln96_kernel<<<(TOK * 32) / 256, 256, 0, stream>>>(
        t96, lng[br], lnb[br], fout[br], nullptr, TOK);
  }

  // cosine fusion -> cat (f16)
  fuse_kernel<<<(TOK * 32) / 256, 256, 0, stream>>>(f1, f2, cath, TOK);
  // im2col for 3x3 SAME conv
  im2col_kernel<<<(TOK * 1728) / 256, 256, 0, stream>>>(
      cath, colh, TOK * 1728);
  // conv as GEMM: [TOK,1728] @ [1728,96] + bf, GELU -> fc (f16)
  gemm16_kernel<<<dim3(CDIM / 32, TOK / 256), 256, 0, stream>>>(
      colh, 1728, Wfh, CDIM, nullptr, fch, CDIM,
      bf, nullptr, TOK, CDIM, 1728, 1);
  // MLP: fc @ W1 + b1, GELU -> hidden (f16)
  gemm16_kernel<<<dim3(384 / 32, TOK / 256), 256, 0, stream>>>(
      fch, CDIM, W1h, 384, nullptr, hidh, 384,
      b1, nullptr, TOK, 384, CDIM, 1);
  // hidden @ W2 + b2 + residual(x) -> output (f32)
  gemm16_kernel<<<dim3(CDIM / 32, TOK / 256), 256, 0, stream>>>(
      hidh, 384, W2h, CDIM, (float*)d_out, nullptr, CDIM,
      b2, x, TOK, CDIM, 384, 0);
}
